// Attention_51823075393746
// MI455X (gfx1250) — compile-verified, matrix operand order checked by reading, not run on hardware
//
#include <hip/hip_runtime.h>
#include <hip/hip_bf16.h>

// ---------------- types ----------------
typedef __attribute__((ext_vector_type(16))) __bf16 bf16x16;
typedef __attribute__((ext_vector_type(8)))  float  f32x8;

union Frag16 {            // 16 bf16 values = A or B fragment of 16x16x32 bf16 WMMA
    unsigned int u[8];
    __bf16       h[16];   // h[2j],h[2j+1] = low/high halves of u[j]
    bf16x16      v;
};
union FragC {             // 16x16 f32 C/D fragment (8 f32 per lane)
    f32x8 v;
    float f[8];
};
union Half8 {             // 8 packed bf16 for 16B stores
    __bf16 h[8];
    uint4  q;
};

// ---------------- constants ----------------
constexpr int B   = 16;
constexpr int C   = 128;
constexpr int HH  = 64;
constexpr int WW  = 64;
constexpr int N   = HH * WW;        // 4096
constexpr int M   = N / 4;          // 1024 pooled
constexpr int C8  = 16;
constexpr int C2  = 64;

// ws element counts (bf16)
constexpr size_t THETA_ELEMS = (size_t)B * N * C8;   // [b][n][c]
constexpr size_t PHI_ELEMS   = (size_t)B * M * C8;   // [b][m][c]
constexpr size_t G_ELEMS     = (size_t)B * C2 * M;   // [b][c2][m]
constexpr size_t O_ELEMS     = (size_t)B * N * C2;   // [b][n][c2]

// ---------------- helpers ----------------
static __device__ __forceinline__ float rowred_max(float v) {
    #pragma unroll
    for (int m = 1; m < 16; m <<= 1) v = fmaxf(v, __shfl_xor(v, m, 32));
    return v;   // xor masks 1..8 stay inside each 16-lane half
}
static __device__ __forceinline__ float rowred_sum(float v) {
    #pragma unroll
    for (int m = 1; m < 16; m <<= 1) v += __shfl_xor(v, m, 32);
    return v;
}
static __device__ __forceinline__ FragC wmma_bf16(const Frag16& a, const Frag16& b, const FragC& c) {
    FragC d;
    d.v = __builtin_amdgcn_wmma_f32_16x16x32_bf16(false, a.v, false, b.v, (short)0, c.v, false, false);
    return d;
}
static __device__ __forceinline__ void ld16B(Frag16& f, int base, const __bf16* p) {
    uint4 t = *reinterpret_cast<const uint4*>(p);
    f.u[base + 0] = t.x; f.u[base + 1] = t.y; f.u[base + 2] = t.z; f.u[base + 3] = t.w;
}

// A-fragment of an f32 weight matrix W[? rows x ldw cols], K-chunk at k0 (32 wide).
// lanes 0-15: M=lane, K pairs {kh+0..7, 16+kh..}; lanes 16-31: same rows, kh=8.
static __device__ __forceinline__ Frag16 ld_wfragA(const float* __restrict__ w,
                                                   int row0, int ldw, int lane, int k0) {
    Frag16 f;
    int mrow = row0 + (lane & 15);
    int kh   = (lane < 16) ? 0 : 8;
    const float* r = w + (size_t)mrow * ldw + k0;
    #pragma unroll
    for (int j = 0; j < 8; ++j) {
        f.h[j]     = (__bf16)r[kh + j];        // u[0..3] <- K kh..kh+7
        f.h[8 + j] = (__bf16)r[16 + kh + j];   // u[4..7] <- K 16+kh..
    }
    return f;
}
// B-fragment of X (K=channel chunk x N=16 pixels), x layout [c][pixel] with row stride N.
static __device__ __forceinline__ Frag16 ld_xfragB(const float* __restrict__ xb,
                                                   int lane, int k0, int p0) {
    Frag16 f;
    int p  = p0 + (lane & 15);
    int kh = (lane < 16) ? 0 : 8;
    const float* col = xb + p + (size_t)k0 * N;
    #pragma unroll
    for (int j = 0; j < 8; ++j) {
        f.h[j]     = (__bf16)col[(size_t)(kh + j) * N];
        f.h[8 + j] = (__bf16)col[(size_t)(16 + kh + j) * N];
    }
    return f;
}

// ================= kernel 1: theta projection =================
// theta[b][n][c] (bf16) = (Wtheta[16,128] x X[128,N]) tile per wave
__global__ void __launch_bounds__(128)
k_proj_theta(const float* __restrict__ x, const float* __restrict__ wt,
             __bf16* __restrict__ theta) {
    int lane = threadIdx.x & 31;
    int wv   = blockIdx.x * 4 + (threadIdx.x >> 5);       // 4096 waves
    int b    = wv >> 8;
    int n0   = (wv & 255) << 4;
    const float* xb = x + (size_t)b * C * N;

    FragC acc;
    #pragma unroll
    for (int r = 0; r < 8; ++r) acc.f[r] = 0.f;
    #pragma unroll
    for (int kc = 0; kc < 4; ++kc) {
        Frag16 a  = ld_wfragA(wt, 0, C, lane, kc * 32);
        Frag16 bx = ld_xfragB(xb, lane, kc * 32, n0);
        acc = wmma_bf16(a, bx, acc);
    }
    // D rows = theta channel, cols = pixel -> store c-contiguous per n
    int n  = n0 + (lane & 15);
    int cb = (lane < 16) ? 0 : 8;
    Half8 o;
    #pragma unroll
    for (int j = 0; j < 8; ++j) o.h[j] = (__bf16)acc.f[j];
    *reinterpret_cast<uint4*>(theta + ((size_t)(b * N + n) * C8 + cb)) = o.q;
}

// ================= kernel 2: phi/g projection + 2x2 maxpool =================
// wave handles (b, chan-tile t in [0,5), pooled row h, half wg): 16ch x (2x32) pixels
__global__ void __launch_bounds__(128)
k_proj_pg(const float* __restrict__ x, const float* __restrict__ wphi,
          const float* __restrict__ wg,
          __bf16* __restrict__ phi, __bf16* __restrict__ g) {
    __shared__ __align__(16) float lds[4][16][64];
    int lane = threadIdx.x & 31;
    int wslt = threadIdx.x >> 5;
    int wv   = blockIdx.x * 4 + wslt;                     // 5120 waves
    int wg2  = wv & 1;
    int h    = (wv >> 1) & 31;
    int r5   = wv >> 6;
    int t    = r5 % 5;
    int b    = r5 / 5;

    const float* wsrc = (t == 0) ? wphi : (wg + (size_t)(t - 1) * 16 * C);
    const float* xb   = x + (size_t)b * C * N;
    float (*L)[64] = lds[wslt];

    #pragma unroll
    for (int pt = 0; pt < 4; ++pt) {
        int row  = 2 * h + (pt >> 1);
        int col0 = 32 * wg2 + 16 * (pt & 1);
        int p0   = row * WW + col0;
        FragC acc;
        #pragma unroll
        for (int r = 0; r < 8; ++r) acc.f[r] = 0.f;
        #pragma unroll
        for (int kc = 0; kc < 4; ++kc) {
            Frag16 a  = ld_wfragA(wsrc, 0, C, lane, kc * 32);
            Frag16 bx = ld_xfragB(xb, lane, kc * 32, p0);
            acc = wmma_bf16(a, bx, acc);
        }
        int pxl = (pt >> 1) * 32 + (pt & 1) * 16 + (lane & 15);
        int ch  = (lane < 16) ? 0 : 8;
        #pragma unroll
        for (int r = 0; r < 8; ++r) L[ch + r][pxl] = acc.f[r];
    }
    asm volatile("s_wait_dscnt 0x0" ::: "memory");

    int m0 = h * 32 + wg2 * 16;   // pooled pixel base
    if (t == 0) {                 // phi[b][m][c], c contiguous
        int mj = lane & 15;
        int cb = (lane < 16) ? 0 : 8;
        Half8 o;
        #pragma unroll
        for (int j = 0; j < 8; ++j) {
            const float* q = L[cb + j];
            o.h[j] = (__bf16)fmaxf(fmaxf(q[2 * mj], q[2 * mj + 1]),
                                   fmaxf(q[32 + 2 * mj], q[32 + 2 * mj + 1]));
        }
        *reinterpret_cast<uint4*>(phi + ((size_t)(b * M + m0 + mj) * C8 + cb)) = o.q;
    } else {                      // g[b][c2][m], m contiguous
        int cl = lane & 15;
        int mh = (lane < 16) ? 0 : 8;
        const float* q = L[cl];
        Half8 o;
        #pragma unroll
        for (int j = 0; j < 8; ++j) {
            int jj = mh + j;
            o.h[j] = (__bf16)fmaxf(fmaxf(q[2 * jj], q[2 * jj + 1]),
                                   fmaxf(q[32 + 2 * jj], q[32 + 2 * jj + 1]));
        }
        int c2 = (t - 1) * 16 + cl;
        *reinterpret_cast<uint4*>(g + ((size_t)(b * C2 + c2) * M + m0 + mh)) = o.q;
    }
}

// ================= kernel 3: fused flash attention =================
// per wave: one 16-query tile; stream 32 chunks of 32 keys with online softmax
__global__ void __launch_bounds__(128)
k_attn(const __bf16* __restrict__ theta, const __bf16* __restrict__ phi,
       const __bf16* __restrict__ g, __bf16* __restrict__ o_ws) {
    __shared__ __align__(16) __bf16 pbuf[4][16][32];
    int lane = threadIdx.x & 31;
    int wslt = threadIdx.x >> 5;
    int wv   = blockIdx.x * 4 + wslt;                     // 4096 waves
    int b    = wv >> 8;
    int n0   = (wv & 255) << 4;
    __bf16 (*Lw)[32] = pbuf[wslt];

    int kh = (lane < 16) ? 0 : 8;

    // theta A fragment: rows = query n, K = 16 channels (upper 16 zero-padded)
    Frag16 thA;
    ld16B(thA, 0, theta + ((size_t)(b * N + n0 + (lane & 15)) * C8 + kh));
    thA.u[4] = thA.u[5] = thA.u[6] = thA.u[7] = 0u;

    FragC acc[4];
    #pragma unroll
    for (int t2 = 0; t2 < 4; ++t2) {
        #pragma unroll
        for (int r = 0; r < 8; ++r) acc[t2].f[r] = 0.f;
    }
    float rm[8], rs[8];
    #pragma unroll
    for (int r = 0; r < 8; ++r) { rm[r] = -INFINITY; rs[r] = 0.f; }
    FragC zc;
    #pragma unroll
    for (int r = 0; r < 8; ++r) zc.f[r] = 0.f;

    const __bf16* phib = phi + (size_t)b * M * C8;
    const __bf16* gb   = g   + (size_t)b * C2 * M;

    for (int mt = 0; mt < 32; ++mt) {
        int m0 = mt * 32;
        if (mt + 1 < 32)   // prefetch next g chunk
            __builtin_prefetch(gb + (size_t)(lane & 31) * 2 * M + (m0 + 32), 0, 0);

        // phi B fragments (K = 16 channels zero-padded to 32)
        Frag16 pb0, pb1;
        ld16B(pb0, 0, phib + ((size_t)(m0 + (lane & 15)) * C8 + kh));
        ld16B(pb1, 0, phib + ((size_t)(m0 + 16 + (lane & 15)) * C8 + kh));
        pb0.u[4] = pb0.u[5] = pb0.u[6] = pb0.u[7] = 0u;
        pb1.u[4] = pb1.u[5] = pb1.u[6] = pb1.u[7] = 0u;

        FragC s0 = wmma_bf16(thA, pb0, zc);
        FragC s1 = wmma_bf16(thA, pb1, zc);

        // online softmax per query row
        float sc[8];
        #pragma unroll
        for (int r = 0; r < 8; ++r) {
            float tm   = rowred_max(fmaxf(s0.f[r], s1.f[r]));
            float newm = fmaxf(rm[r], tm);
            sc[r]      = __expf(rm[r] - newm);
            rm[r]      = newm;
            float p0   = __expf(s0.f[r] - newm);
            float p1   = __expf(s1.f[r] - newm);
            rs[r]      = rs[r] * sc[r] + rowred_sum(p0 + p1);
            s0.f[r] = p0; s1.f[r] = p1;
        }
        #pragma unroll
        for (int t2 = 0; t2 < 4; ++t2) {
            #pragma unroll
            for (int r = 0; r < 8; ++r) acc[t2].f[r] *= sc[r];
        }

        // C-layout P -> LDS -> A-layout fragment (16n x 32m)
        int nbase = (lane < 16) ? 0 : 8;
        int col   = lane & 15;
        #pragma unroll
        for (int r = 0; r < 8; ++r) {
            Lw[nbase + r][col]      = (__bf16)s0.f[r];
            Lw[nbase + r][16 + col] = (__bf16)s1.f[r];
        }
        asm volatile("s_wait_dscnt 0x0" ::: "memory");
        Frag16 pA;
        ld16B(pA, 0, &Lw[lane & 15][kh]);
        ld16B(pA, 4, &Lw[lane & 15][16 + kh]);

        // O += P x G  (4 channel tiles of 16)
        #pragma unroll
        for (int t2 = 0; t2 < 4; ++t2) {
            Frag16 gB;
            const __bf16* gp = gb + (size_t)(t2 * 16 + (lane & 15)) * M + m0 + kh;
            ld16B(gB, 0, gp);
            ld16B(gB, 4, gp + 16);
            acc[t2] = wmma_bf16(pA, gB, acc[t2]);
        }
    }

    // finalize: divide by row sums, store o_ws[b][n][c2]
    #pragma unroll
    for (int r = 0; r < 8; ++r) {
        float inv = 1.0f / rs[r];
        int n = n0 + r + ((lane < 16) ? 0 : 8);
        #pragma unroll
        for (int t2 = 0; t2 < 4; ++t2) {
            int c2 = t2 * 16 + (lane & 15);
            o_ws[(size_t)(b * N + n) * C2 + c2] = (__bf16)(acc[t2].f[r] * inv);
        }
    }
}

// ================= kernel 4: output projection + residual =================
__global__ void __launch_bounds__(128)
k_final(const float* __restrict__ x, const float* __restrict__ wo,
        const float* __restrict__ gp, const __bf16* __restrict__ o_ws,
        float* __restrict__ out) {
    int lane = threadIdx.x & 31;
    int wv   = blockIdx.x * 4 + (threadIdx.x >> 5);       // 32768 waves
    int nt   = wv & 255;
    int cot  = (wv >> 8) & 7;
    int b    = wv >> 11;
    int n0   = nt << 4;
    int co0  = cot << 4;
    int kh   = (lane < 16) ? 0 : 8;

    FragC acc;
    #pragma unroll
    for (int r = 0; r < 8; ++r) acc.f[r] = 0.f;

    #pragma unroll
    for (int kc = 0; kc < 2; ++kc) {
        Frag16 a = ld_wfragA(wo, co0, C2, lane, kc * 32);
        Frag16 bo;
        const __bf16* ob =
            o_ws + ((size_t)(b * N + n0 + (lane & 15)) * C2 + kc * 32 + kh);
        ld16B(bo, 0, ob);
        ld16B(bo, 4, ob + 16);
        acc = wmma_bf16(a, bo, acc);
    }

    float gamma = gp[0];
    int n = n0 + (lane & 15);
    #pragma unroll
    for (int r = 0; r < 8; ++r) {
        int co = co0 + r + ((lane < 16) ? 0 : 8);
        size_t idx = ((size_t)b * C + co) * N + n;
        out[idx] = gamma * acc.f[r] + x[idx];
    }
}

// ================= launcher =================
extern "C" void kernel_launch(void* const* d_in, const int* in_sizes, int n_in,
                              void* d_out, int out_size, void* d_ws, size_t ws_size,
                              hipStream_t stream) {
    const float* x      = (const float*)d_in[0];
    const float* wtheta = (const float*)d_in[1];
    const float* wphi   = (const float*)d_in[2];
    const float* wg     = (const float*)d_in[3];
    const float* wo     = (const float*)d_in[4];
    const float* gamma  = (const float*)d_in[5];
    float* out          = (float*)d_out;

    __bf16* theta = (__bf16*)d_ws;
    __bf16* phi   = theta + THETA_ELEMS;
    __bf16* g     = phi + PHI_ELEMS;
    __bf16* o_ws  = g + G_ELEMS;
    // total ws usage: (THETA+PHI+G+O)*2 bytes ~= 12.6 MB

    k_proj_theta<<<1024, 128, 0, stream>>>(x, wtheta, theta);
    k_proj_pg   <<<1280, 128, 0, stream>>>(x, wphi, wg, phi, g);
    k_attn      <<<1024, 128, 0, stream>>>(theta, phi, g, o_ws);
    k_final     <<<8192, 128, 0, stream>>>(x, wo, gamma, o_ws, out);
}
